// Attention_75110388072976
// MI455X (gfx1250) — matrix operand order repaired and checked by reading, not compiled, not run on hardware
//
#include <hip/hip_runtime.h>

// ---------------------------------------------------------------------------
// Attention forward for gfx1250 (MI455X, wave32, WMMA).
//   scores = QK^T/8 + log(p+1e-12); mask==0 -> -1e9; softmax; out = P V
// Memory-bound: ~0.8 GB mandatory HBM traffic @ 23.3 TB/s => ~34 us floor.
// One wave per 16-query tile; unnormalized exp(s) staged in 64 KB LDS (f16),
// consumed directly by the PV v_wmma_f32_16x16x32_f16 chain.
// Softmax runs in the exp2 domain on raw v_log_f32/v_exp_f32 to keep VALU
// off the critical path; p/mask/p_attn use non-temporal cache policy so the
// reusable Q/K/V panels stay L2-resident.
// ---------------------------------------------------------------------------

typedef __attribute__((ext_vector_type(16))) _Float16 v16h;
typedef __attribute__((ext_vector_type(8)))  float    v8f;
typedef __attribute__((ext_vector_type(4)))  _Float16 h4;
typedef __attribute__((ext_vector_type(4)))  float    f4;

static constexpr int Bb = 2, Hh = 8, Nn = 2048, Dd = 64;

// 0.125 / ln(2): converts qk/sqrt(64) into the exp2 domain
#define SCALE_LOG2E 0.18033688011112043f

static __device__ __forceinline__ v8f wmma_f16(v16h a, v16h b, v8f c) {
    // (neg_a, A, neg_b, B, c_mod, C, reuse_a, reuse_b)
    return __builtin_amdgcn_wmma_f32_16x16x32_f16(false, a, false, b,
                                                  (short)0, c, false, false);
}

// Raw hardware transcendentals: v_log_f32 = log2, v_exp_f32 = exp2.
// Inputs here are always normal-range, so skip the libm denorm fixups.
// Fallbacks use standard exp2f/log2f (declared for both host and device
// passes -- the host pass type-checks these bodies too).
static __device__ __forceinline__ float fast_log2(float x) {
#if __has_builtin(__builtin_amdgcn_logf)
    return __builtin_amdgcn_logf(x);
#else
    return log2f(x);
#endif
}
static __device__ __forceinline__ float fast_exp2(float x) {
#if __has_builtin(__builtin_amdgcn_exp2f)
    return __builtin_amdgcn_exp2f(x);
#else
    return exp2f(x);
#endif
}

// A-fragment (16x32 f16) element i -> K index for this lane half (hi = lane>>4)
static __device__ __forceinline__ int a_kidx(int i, int hi) {
    return (i < 8) ? (hi * 8 + i) : (16 + hi * 8 + (i - 8));
}

__global__ __launch_bounds__(32)
void attn_fwd_kernel(const float* __restrict__ Q,
                     const float* __restrict__ K,
                     const float* __restrict__ V,
                     const int*   __restrict__ M,
                     const float* __restrict__ P,
                     float* __restrict__ out,
                     float* __restrict__ pattn)
{
    __shared__ _Float16 lds_e[16 * Nn];   // 64 KB: unnormalized exp(scores)

    const int lane = threadIdx.x;         // 0..31 (wave32)
    const int lo   = lane & 15;
    const int hi   = lane >> 4;

    const int bid = blockIdx.x;           // B*H*(N/16) = 2048 blocks
    const int qt  = bid & (Nn / 16 - 1);  // query tile within (b,h)
    const int bh  = bid >> 7;             // N/16 = 128 = 2^7
    const int q0  = qt * 16;

    const size_t qkv_base = (size_t)bh * Nn * Dd;
    const size_t pm_base  = (size_t)bh * Nn * Nn + (size_t)q0 * Nn;

    // ---- Q A-fragments (rows q0+lo, K-dim 0..31 and 32..63), f32 -> f16 ----
    v16h a0, a1;
    {
        const float* qrow = Q + qkv_base + (size_t)(q0 + lo) * Dd;
        #pragma unroll
        for (int i = 0; i < 16; i += 2) {
            const int k = a_kidx(i, hi);          // k even; pairs contiguous
            a0[i]     = (_Float16)qrow[k];
            a0[i + 1] = (_Float16)qrow[k + 1];
            a1[i]     = (_Float16)qrow[32 + k];
            a1[i + 1] = (_Float16)qrow[32 + k + 1];
        }
    }

    float lsum[8];
    #pragma unroll
    for (int r = 0; r < 8; ++r) lsum[r] = 0.0f;

    // ---- Pass 1: S = QK^T/8 + log(p), mask, e = exp(S) -> LDS; row sums ----
    for (int kc = 0; kc < Nn / 16; ++kc) {
        // B-fragments of K^T: lane column = key kc*16+lo; K-dim = a_kidx(i, hi)
        const float* krow = K + qkv_base + (size_t)(kc * 16 + lo) * Dd;
        v16h b0, b1;
        #pragma unroll
        for (int i = 0; i < 16; ++i) {
            b0[i] = (_Float16)krow[a_kidx(i, hi)];        // dims 0..31
            b1[i] = (_Float16)krow[32 + a_kidx(i, hi)];   // dims 32..63
        }

        v8f c = {};
        c = wmma_f16(a0, b0, c);
        c = wmma_f16(a1, b1, c);              // v_wmma_f32_16x16x32_f16

        const int col = kc * 16 + lo;
        #pragma unroll
        for (int r = 0; r < 8; ++r) {
            const int    row = r + 8 * hi;    // C/D layout: M = r + 8*(lane>>4)
            const size_t idx = pm_base + (size_t)row * Nn + col;
            // Streaming, touch-once tensors: non-temporal loads (TH=NT)
            const float pv = __builtin_nontemporal_load(&P[idx]);
            const int   mv = __builtin_nontemporal_load(&M[idx]);
            // exp2 domain: exp(s) == exp2(c*0.125*log2(e) + log2(p+eps))
            float s2 = fmaf(c[r], SCALE_LOG2E, fast_log2(pv + 1e-12f));
            s2 = (mv == 0) ? -1e9f : s2;      // exp2(-1e9) == 0.0f
            const float e = fast_exp2(s2);
            lsum[r] += e;
            lds_e[row * Nn + col] = (_Float16)e;
        }
    }
    __syncthreads();

    // ---- Row-sum reduction across the 16 lanes of each half-wave group ----
    float linv[8];
    #pragma unroll
    for (int r = 0; r < 8; ++r) {
        float v = lsum[r];
        #pragma unroll
        for (int off = 8; off >= 1; off >>= 1)
            v += __shfl_xor(v, off, 16);      // segment width 16
        linv[r] = 1.0f / v;                   // rows r+8*hi fully summed
    }

    // ---- Pass 2: out_tile = P_tile x V (f16 WMMA, 32 keys / iter) ----
    v8f acc0 = {}, acc1 = {}, acc2 = {}, acc3 = {};
    for (int kc = 0; kc < Nn / 32; ++kc) {
        // A-fragment of probabilities from LDS (row = lo)
        v16h pa;
        #pragma unroll
        for (int i = 0; i < 16; i += 2) {
            const int k = kc * 32 + a_kidx(i, hi);
            pa[i]     = lds_e[lo * Nn + k];
            pa[i + 1] = lds_e[lo * Nn + k + 1];
        }
        // B-fragments of V for the four 16-wide D slices
        v16h vb0, vb1, vb2, vb3;
        #pragma unroll
        for (int i = 0; i < 16; ++i) {
            const float* vrow = V + qkv_base + (size_t)(kc * 32 + a_kidx(i, hi)) * Dd + lo;
            vb0[i] = (_Float16)vrow[0];
            vb1[i] = (_Float16)vrow[16];
            vb2[i] = (_Float16)vrow[32];
            vb3[i] = (_Float16)vrow[48];
        }
        acc0 = wmma_f16(pa, vb0, acc0);
        acc1 = wmma_f16(pa, vb1, acc1);
        acc2 = wmma_f16(pa, vb2, acc2);
        acc3 = wmma_f16(pa, vb3, acc3);
    }

    // ---- out = acc / l  (lane already holds 1/l for its own rows) ----
    #pragma unroll
    for (int r = 0; r < 8; ++r) {
        const int    row = r + 8 * hi;
        const size_t ob  = qkv_base + (size_t)(q0 + row) * Dd + lo;
        const float  inv = linv[r];
        out[ob +  0] = acc0[r] * inv;
        out[ob + 16] = acc1[r] * inv;
        out[ob + 32] = acc2[r] * inv;
        out[ob + 48] = acc3[r] * inv;
    }

    // ---- Broadcast 1/l for all 16 rows to every lane (compile-time index) --
    float invall[16];
    #pragma unroll
    for (int row = 0; row < 16; ++row)
        invall[row] = __shfl(linv[row & 7], (row >> 3) * 16, 32);

    // ---- Stream p_attn = e / l from LDS -> global (NT float4 stores) ----
    #pragma unroll
    for (int row = 0; row < 16; ++row) {
        const float inv  = invall[row];
        float* prow = pattn + pm_base + (size_t)row * Nn;
        for (int c = lane * 4; c < Nn; c += 32 * 4) {
            const h4 ev = *(const h4*)&lds_e[row * Nn + c];  // 8B-aligned
            f4 o;
            o.x = (float)ev.x * inv;
            o.y = (float)ev.y * inv;
            o.z = (float)ev.z * inv;
            o.w = (float)ev.w * inv;
            __builtin_nontemporal_store(o, (f4*)&prow[c]);   // 16B-aligned
        }
    }
}

extern "C" void kernel_launch(void* const* d_in, const int* in_sizes, int n_in,
                              void* d_out, int out_size, void* d_ws, size_t ws_size,
                              hipStream_t stream) {
    (void)in_sizes; (void)n_in; (void)d_ws; (void)ws_size; (void)out_size;
    const float* Q = (const float*)d_in[0];
    const float* K = (const float*)d_in[1];
    const float* V = (const float*)d_in[2];
    const int*   M = (const int*)  d_in[3];
    const float* P = (const float*)d_in[4];

    float* out   = (float*)d_out;                       // [B,H,N,D]
    float* pattn = out + (size_t)Bb * Hh * Nn * Dd;     // [B,H,N,N]

    const int blocks = Bb * Hh * (Nn / 16);             // 2048 waves
    attn_fwd_kernel<<<dim3(blocks), dim3(32), 0, stream>>>(Q, K, V, M, P, out, pattn);
}
